// PointTransformerCls_24180665877119
// MI455X (gfx1250) — matrix operand order repaired and checked by reading, not compiled, not run on hardware
//
#include <hip/hip_runtime.h>
#include <math.h>

// ---------------------------------------------------------------------------
// Point Transformer classifier forward for gfx1250 (MI455X).
// Linear layers: bf16 WMMA GEMM (v_wmma_f32_16x16x32_bf16), fp32 accumulate.
// A and W are pre-converted to zero-padded bf16 (W transposed) so the GEMM
// hot loop is pure unconditional global_load_b128 + v_wmma: no LDS staging,
// no barriers, no exec-mask branching.
// ---------------------------------------------------------------------------

typedef __attribute__((ext_vector_type(16))) __bf16 bf16x16;
typedef __attribute__((ext_vector_type(8)))  __bf16 bf16x8;
typedef __attribute__((ext_vector_type(8)))  float  f32x8;

#define GEMM_WAVES 8       // waves per block; block covers 128(M) x 16(N)

// ======================= WMMA GEMM =======================
// C[M,N] = act(Ab[M,Kp] @ Wt[Np,Kp]^T + bias) [+ resid]
// Ab: bf16 row-major, K padded to Kp (mult of 32), zero-filled.
// Wt: bf16, TRANSPOSED weights [n][k], N padded to Np (mult of 16), zero-filled.
__global__ void __launch_bounds__(256)
gemm_bf16_wmma_kernel(const __bf16* __restrict__ Ab, const __bf16* __restrict__ Wt,
                      const float* __restrict__ bias, const float* __restrict__ resid,
                      float* __restrict__ C, int M, int N, int Kp, int relu)
{
    const int lane  = threadIdx.x & 31;
    const int wave  = threadIdx.x >> 5;
    const int half  = lane >> 4;             // 0: lanes 0-15, 1: lanes 16-31
    const int l15   = lane & 15;
    const int nbase = blockIdx.x * 16;
    const int mbase = (blockIdx.y * GEMM_WAVES + wave) * 16;
    const int arow  = mbase + l15;
    const int arowc = arow < M ? arow : (M - 1);   // clamp: OOB rows never stored

    // Per-lane base pointers (A: 16-bit A 16x32 layout; B: col = lane&15).
    //  A lane reads k in {half*8+[0,8)} and {16+half*8+[0,8)}
    //  B lane reads k in {half*8+[0,8)} and {16+half*8+[0,8)}  (same as A)
    const __bf16* __restrict__ Arow = Ab + (size_t)arowc * Kp + half * 8;
    const __bf16* __restrict__ Brow = Wt + (size_t)(nbase + l15) * Kp + half * 8;

    f32x8 acc = {0.f, 0.f, 0.f, 0.f, 0.f, 0.f, 0.f, 0.f};

    for (int k0 = 0; k0 < Kp; k0 += 32) {
        bf16x8 a0 = *(const bf16x8*)(Arow + k0);
        bf16x8 a1 = *(const bf16x8*)(Arow + k0 + 16);
        bf16x8 b0 = *(const bf16x8*)(Brow + k0);
        bf16x8 b1 = *(const bf16x8*)(Brow + k0 + 16);

        if (k0 + 32 < Kp) {
            __builtin_prefetch(Arow + k0 + 32, 0, 1);
            __builtin_prefetch(Brow + k0 + 32, 0, 1);
        }

        bf16x16 af  = __builtin_shufflevector(a0, a1, 0, 1, 2, 3, 4, 5, 6, 7,
                                              8, 9, 10, 11, 12, 13, 14, 15);
        bf16x16 bfr = __builtin_shufflevector(b0, b1, 0, 1, 2, 3, 4, 5, 6, 7,
                                              8, 9, 10, 11, 12, 13, 14, 15);

        acc = __builtin_amdgcn_wmma_f32_16x16x32_bf16(false, af, false, bfr,
                                                      (short)0, acc, false, false);
    }

    // C/D layout: VGPR r, lanes 0-15 -> M=r; lanes 16-31 -> M=r+8; N = lane&15.
    const int n   = nbase + l15;
    const int nok = n < N;
    const float bv = (bias && nok) ? bias[n] : 0.f;
#pragma unroll
    for (int r = 0; r < 8; ++r) {
        int m = mbase + r + 8 * half;
        if (m < M && nok) {
            float v = acc[r] + bv;
            if (relu)  v = fmaxf(v, 0.f);
            if (resid) v += resid[(size_t)m * N + n];
            C[(size_t)m * N + n] = v;
        }
    }
}

// A f32 [M,K] -> bf16 [M,Kp], zero pad
__global__ void a_to_bf16_kernel(const float* __restrict__ A, __bf16* __restrict__ Ab,
                                 long long M, int K, int Kp)
{
    long long t = (long long)blockIdx.x * blockDim.x + threadIdx.x;
    long long total = M * Kp;
    if (t >= total) return;
    int k = (int)(t % Kp); long long m = t / Kp;
    float v = (k < K) ? A[m * K + k] : 0.f;
    Ab[t] = (__bf16)v;
}

// W f32 [K,N] -> bf16 transposed [Np,Kp], zero pad
__global__ void w_to_bf16t_kernel(const float* __restrict__ W, __bf16* __restrict__ Wt,
                                  int K, int N, int Kp, int Np)
{
    long long t = (long long)blockIdx.x * blockDim.x + threadIdx.x;
    long long total = (long long)Np * Kp;
    if (t >= total) return;
    int k = (int)(t % Kp); int n = (int)(t / Kp);
    float v = (k < K && n < N) ? W[(size_t)k * N + n] : 0.f;
    Wt[t] = (__bf16)v;
}

// ======================= helpers / pointwise =======================

// x [B,3,N] -> xyz [B,N,3]
__global__ void xyz_transpose_kernel(const float* __restrict__ x, float* __restrict__ xyz,
                                     int B, int N)
{
    long long t = (long long)blockIdx.x * blockDim.x + threadIdx.x;
    long long total = (long long)B * N * 3;
    if (t >= total) return;
    int c = (int)(t % 3); long long bn = t / 3;
    int n = (int)(bn % N); int b = (int)(bn / N);
    xyz[t] = x[((size_t)b * 3 + c) * N + n];
}

// stable top-Kn nearest neighbors (insertion sort => argsort tie order)
__global__ void knn_kernel(const float* __restrict__ q, const float* __restrict__ pts,
                           int* __restrict__ out, int B, int Nq, int Np, int Kn)
{
    int t = blockIdx.x * blockDim.x + threadIdx.x;
    if (t >= B * Nq) return;
    int b = t / Nq, n = t - b * Nq;
    const float* qp = q + ((size_t)b * Nq + n) * 3;
    float qx = qp[0], qy = qp[1], qz = qp[2];
    const float* base = pts + (size_t)b * Np * 3;
    float bd[16]; int bi[16];
    for (int j = 0; j < 16; ++j) { bd[j] = 3.4e38f; bi[j] = 0; }
    for (int m = 0; m < Np; ++m) {
        float dx = base[3 * m] - qx, dy = base[3 * m + 1] - qy, dz = base[3 * m + 2] - qz;
        float d = dx * dx + dy * dy + dz * dz;
        if (d < bd[Kn - 1]) {
            int j = Kn - 1;
            while (j > 0 && bd[j - 1] > d) { bd[j] = bd[j - 1]; bi[j] = bi[j - 1]; --j; }
            bd[j] = d; bi[j] = m;
        }
    }
    int* o = out + (size_t)t * Kn;
    for (int j = 0; j < Kn; ++j) o[j] = bi[j];
}

// farthest point sampling; one block per batch; deterministic start at 0
__global__ void fps_kernel(const float* __restrict__ xyz, int* __restrict__ idx_out,
                           int N, int npoint)
{
    __shared__ float dist[1024];
    __shared__ float rv[256];
    __shared__ int   ri[256];
    __shared__ int   cur;
    int b = blockIdx.x;
    const float* base = xyz + (size_t)b * N * 3;
    for (int i = threadIdx.x; i < N; i += blockDim.x) dist[i] = 1e10f;
    if (threadIdx.x == 0) cur = 0;
    __syncthreads();
    for (int s = 0; s < npoint; ++s) {
        int far = cur;
        if (threadIdx.x == 0) idx_out[b * npoint + s] = far;
        float cx = base[3 * far], cy = base[3 * far + 1], cz = base[3 * far + 2];
        float bv = -1.f; int bi = 0x7fffffff;
        for (int i = threadIdx.x; i < N; i += blockDim.x) {
            float dx = base[3 * i] - cx, dy = base[3 * i + 1] - cy, dz = base[3 * i + 2] - cz;
            float d = dx * dx + dy * dy + dz * dz;
            float nd = fminf(dist[i], d);
            dist[i] = nd;
            if (nd > bv || (nd == bv && i < bi)) { bv = nd; bi = i; }
        }
        rv[threadIdx.x] = bv; ri[threadIdx.x] = bi;
        __syncthreads();
        for (int off = blockDim.x >> 1; off > 0; off >>= 1) {
            if ((int)threadIdx.x < off) {
                if (rv[threadIdx.x + off] > rv[threadIdx.x] ||
                    (rv[threadIdx.x + off] == rv[threadIdx.x] &&
                     ri[threadIdx.x + off] < ri[threadIdx.x])) {
                    rv[threadIdx.x] = rv[threadIdx.x + off];
                    ri[threadIdx.x] = ri[threadIdx.x + off];
                }
            }
            __syncthreads();
        }
        if (threadIdx.x == 0) cur = ri[0];
        __syncthreads();
    }
}

// dst[b,l,:] = src[b, idx[b,l], :]   (idx flat length L per batch)
__global__ void gather_kernel(const float* __restrict__ src, const int* __restrict__ idx,
                              float* __restrict__ dst, int B, int N, int L, int C)
{
    long long t = (long long)blockIdx.x * blockDim.x + threadIdx.x;
    long long total = (long long)B * L * C;
    if (t >= total) return;
    int c = (int)(t % C); long long bl = t / C;
    int l = (int)(bl % L); int b = (int)(bl / L);
    int m = idx[(size_t)b * L + l];
    dst[t] = src[((size_t)b * N + m) * C + c];
}

// rel[b,n,j,:] = xyz[b,n,:] - xyz[b,knn[b,n,j],:]
__global__ void rel_kernel(const float* __restrict__ xyz, const int* __restrict__ knn,
                           float* __restrict__ rel, int B, int N, int Kn)
{
    long long t = (long long)blockIdx.x * blockDim.x + threadIdx.x;
    long long total = (long long)B * N * Kn;
    if (t >= total) return;
    long long bn = t / Kn;
    int n = (int)(bn % N); int b = (int)(bn / N);
    int m = knn[t];
    const float* pq = xyz + ((size_t)b * N + n) * 3;
    const float* pm = xyz + ((size_t)b * N + m) * 3;
    float* r = rel + (size_t)t * 3;
    r[0] = pq[0] - pm[0]; r[1] = pq[1] - pm[1]; r[2] = pq[2] - pm[2];
}

// out[b,n,j,c] = q[b,n,c] - kf[b,knn[b,n,j],c] + pos[b,n,j,c]
__global__ void attn_in_kernel(const float* __restrict__ q, const float* __restrict__ kf,
                               const int* __restrict__ knn, const float* __restrict__ pos,
                               float* __restrict__ out, int B, int N, int Kn, int D)
{
    long long t = (long long)blockIdx.x * blockDim.x + threadIdx.x;
    long long total = (long long)B * N * Kn * D;
    if (t >= total) return;
    int c = (int)(t % D); long long r = t / D;
    long long bn = r / Kn;
    int n = (int)(bn % N); int b = (int)(bn / N);
    int m = knn[r];
    out[t] = q[((size_t)b * N + n) * D + c] - kf[((size_t)b * N + m) * D + c] + pos[t];
}

// out[b,n,j,c] = vf[b,knn[b,n,j],c] + pos[b,n,j,c]
__global__ void vpos_kernel(const float* __restrict__ vf, const int* __restrict__ knn,
                            const float* __restrict__ pos, float* __restrict__ out,
                            int B, int N, int Kn, int D)
{
    long long t = (long long)blockIdx.x * blockDim.x + threadIdx.x;
    long long total = (long long)B * N * Kn * D;
    if (t >= total) return;
    int c = (int)(t % D); long long r = t / D;
    long long bn = r / Kn;
    int b = (int)(bn / N);
    int m = knn[r];
    out[t] = vf[((size_t)b * ((size_t)N) + m) * D + c] + pos[t];
}

// softmax over the k axis then weighted sum: res[b,n,c] = sum_j softmax_j(attn*scale) * vpos
__global__ void attn_apply_kernel(const float* __restrict__ attn, const float* __restrict__ vpos,
                                  float* __restrict__ out, int B, int N, int Kn, int D, float scale)
{
    long long t = (long long)blockIdx.x * blockDim.x + threadIdx.x;
    long long total = (long long)B * N * D;
    if (t >= total) return;
    int c = (int)(t % D); long long bn = t / D;
    const float* ap = attn + bn * (size_t)Kn * D + c;
    const float* vp = vpos + bn * (size_t)Kn * D + c;
    float av[16];
    float mx = -3.4e38f;
    for (int j = 0; j < Kn; ++j) { av[j] = ap[(size_t)j * D] * scale; mx = fmaxf(mx, av[j]); }
    float s = 0.f, acc = 0.f;
    for (int j = 0; j < Kn; ++j) {
        float e = expf(av[j] - mx);
        s += e;
        acc += e * vp[(size_t)j * D];
    }
    out[t] = acc / s;
}

// out[b,s,c] = max_j g[b,s,j,c]
__global__ void maxk_kernel(const float* __restrict__ g, float* __restrict__ out,
                            int B, int S, int Kn, int C)
{
    long long t = (long long)blockIdx.x * blockDim.x + threadIdx.x;
    long long total = (long long)B * S * C;
    if (t >= total) return;
    int c = (int)(t % C); long long bs = t / C;
    const float* p = g + bs * (size_t)Kn * C + c;
    float m = p[0];
    for (int j = 1; j < Kn; ++j) m = fmaxf(m, p[(size_t)j * C]);
    out[t] = m;
}

// out[b,c] = mean_n f[b,n,c]
__global__ void mean_kernel(const float* __restrict__ f, float* __restrict__ out,
                            int B, int N, int C)
{
    int t = blockIdx.x * blockDim.x + threadIdx.x;
    if (t >= B * C) return;
    int c = t % C, b = t / C;
    float s = 0.f;
    for (int n = 0; n < N; ++n) s += f[((size_t)b * N + n) * C + c];
    out[t] = s / (float)N;
}

// in-place eval-mode batchnorm + relu
__global__ void bn_relu_kernel(float* __restrict__ x, const float* __restrict__ g,
                               const float* __restrict__ b, const float* __restrict__ rm,
                               const float* __restrict__ rv, int M, int C)
{
    int t = blockIdx.x * blockDim.x + threadIdx.x;
    if (t >= M * C) return;
    int c = t % C;
    float v = (x[t] - rm[c]) * rsqrtf(rv[c] + 1e-5f) * g[c] + b[c];
    x[t] = fmaxf(v, 0.f);
}

// ======================= host-side orchestration =======================

namespace {

struct Lin { const float* w; const float* b; };
struct TBp { Lin fc1, fc2, d0, d1, g0, g1; const float *wq, *wk, *wv; };
struct BNp { const float *g, *b, *rm, *rv; };

struct Bump {
    char* base; size_t off;
    float*  f(size_t n) { float*  r = (float*)(base + off);  off = (off + n * 4 + 255) & ~(size_t)255; return r; }
    int*    i(size_t n) { int*    r = (int*)(base + off);    off = (off + n * 4 + 255) & ~(size_t)255; return r; }
    __bf16* h(size_t n) { __bf16* r = (__bf16*)(base + off); off = (off + n * 2 + 255) & ~(size_t)255; return r; }
};

inline unsigned cdiv(long long a, long long b) { return (unsigned)((a + b - 1) / b); }

// full GEMM: convert A -> padded bf16, W -> transposed padded bf16, run WMMA kernel
inline void gemm(hipStream_t s, Bump& ws, const float* A, const float* W, const float* bias,
                 const float* resid, float* C, long long M, int N, int K, int relu)
{
    int Kp = (K + 31) & ~31;
    int Np = (N + 15) & ~15;
    size_t mark = ws.off;
    __bf16* Ab = ws.h((size_t)M * Kp);
    __bf16* Wt = ws.h((size_t)Np * Kp);
    a_to_bf16_kernel<<<cdiv(M * (long long)Kp, 256), 256, 0, s>>>(A, Ab, M, K, Kp);
    w_to_bf16t_kernel<<<cdiv((long long)Np * Kp, 256), 256, 0, s>>>(W, Wt, K, N, Kp, Np);
    dim3 grid(cdiv(N, 16), cdiv(M, 16 * GEMM_WAVES));
    gemm_bf16_wmma_kernel<<<grid, 256, 0, s>>>(Ab, Wt, bias, resid, C, (int)M, N, Kp, relu);
    ws.off = mark;   // stream-ordered reuse is safe
}

} // namespace

extern "C" void kernel_launch(void* const* d_in, const int* in_sizes, int n_in,
                              void* d_out, int out_size, void* d_ws, size_t ws_size,
                              hipStream_t stream)
{
    (void)in_sizes; (void)n_in; (void)out_size; (void)ws_size;
    int p = 0;
    auto nxt  = [&]() { return (const float*)d_in[p++]; };
    auto rLin = [&]() { Lin l; l.w = nxt(); l.b = nxt(); return l; };
    auto rTB  = [&]() { TBp t; t.fc1 = rLin(); t.fc2 = rLin(); t.d0 = rLin(); t.d1 = rLin();
                        t.g0 = rLin(); t.g1 = rLin(); t.wq = nxt(); t.wk = nxt(); t.wv = nxt(); return t; };
    auto rBN  = [&]() { BNp b; b.g = nxt(); b.b = nxt(); b.rm = nxt(); b.rv = nxt(); return b; };

    // ---- inputs in setup_inputs() insertion order ----
    const float* X = nxt();                 // x [8,3,1024]
    Lin f1a = rLin(), f1b = rLin();         // params['fc1']
    TBp t1 = rTB(); Lin td1a = rLin(), td1b = rLin();
    TBp t2 = rTB(); Lin td2a = rLin(), td2b = rLin();
    TBp t3 = rTB(); Lin td3a = rLin(), td3b = rLin();
    TBp t4 = rTB(); Lin td4a = rLin(), td4b = rLin();
    TBp t5 = rTB();
    Lin hl1 = rLin(); BNp hb1 = rBN();
    Lin hl2 = rLin(); BNp hb2 = rBN();
    Lin hl3 = rLin();

    const int Bb = 8;
    const float SCALE = 1.0f / sqrtf(512.0f);
    Bump ws{(char*)d_ws, 0};

    auto L1 = [&](long long n) { return dim3(cdiv(n, 256)); };

    // ---- transformer block: feats updated in place ----
    auto tblock = [&](const TBp& t, const float* xyz, float* feats, int N, int dp) {
        size_t mark = ws.off;
        int Kn  = N < 16 ? N : 16;
        long long BN = (long long)Bb * N, BNK = BN * Kn;

        int* knn = ws.i(BNK);
        knn_kernel<<<L1(BN), 256, 0, stream>>>(xyz, xyz, knn, Bb, N, N, Kn);

        float* x512 = ws.f(BN * 512);
        gemm(stream, ws, feats, t.fc1.w, t.fc1.b, nullptr, x512, BN, 512, dp, 0);
        float* q  = ws.f(BN * 512);
        float* kf = ws.f(BN * 512);
        float* vf = ws.f(BN * 512);
        gemm(stream, ws, x512, t.wq, nullptr, nullptr, q,  BN, 512, 512, 0);
        gemm(stream, ws, x512, t.wk, nullptr, nullptr, kf, BN, 512, 512, 0);
        gemm(stream, ws, x512, t.wv, nullptr, nullptr, vf, BN, 512, 512, 0);

        float* rel = ws.f(BNK * 3);
        rel_kernel<<<L1(BNK), 256, 0, stream>>>(xyz, knn, rel, Bb, N, Kn);

        float* posh = ws.f(BNK * 512);
        float* pos  = ws.f(BNK * 512);
        gemm(stream, ws, rel,  t.d0.w, t.d0.b, nullptr, posh, BNK, 512, 3,   1);
        gemm(stream, ws, posh, t.d1.w, t.d1.b, nullptr, pos,  BNK, 512, 512, 0);

        float* ain = ws.f(BNK * 512);
        attn_in_kernel<<<L1(BNK * 512), 256, 0, stream>>>(q, kf, knn, pos, ain, Bb, N, Kn, 512);

        // fc_gamma MLP: hidden reuses posh, attn output reuses ain
        gemm(stream, ws, ain,  t.g0.w, t.g0.b, nullptr, posh, BNK, 512, 512, 1);
        gemm(stream, ws, posh, t.g1.w, t.g1.b, nullptr, ain,  BNK, 512, 512, 0);

        float* vpos = posh;   // posh free now
        vpos_kernel<<<L1(BNK * 512), 256, 0, stream>>>(vf, knn, pos, vpos, Bb, N, Kn, 512);

        float* res = q;       // q free now
        attn_apply_kernel<<<L1(BN * 512), 256, 0, stream>>>(ain, vpos, res, Bb, N, Kn, 512, SCALE);

        // fc2 + residual, in place on feats
        gemm(stream, ws, res, t.fc2.w, t.fc2.b, feats, feats, BN, dp, 512, 0);
        ws.off = mark;
    };

    // ---- transition down: returns new xyz/feats (persistent allocs) ----
    auto tdown = [&](Lin a, Lin b2, const float* xyz, const float* feats, int N, int C,
                     float*& nxyz, float*& nfeats) {
        int S = N / 4, C2 = 2 * C;
        int Kn = N < 16 ? N : 16;
        nxyz   = ws.f((size_t)Bb * S * 3);
        nfeats = ws.f((size_t)Bb * S * C2);
        size_t mark = ws.off;

        int* fidx = ws.i((size_t)Bb * S);
        fps_kernel<<<Bb, 256, 0, stream>>>(xyz, fidx, N, S);
        gather_kernel<<<L1((long long)Bb * S * 3), 256, 0, stream>>>(xyz, fidx, nxyz, Bb, N, S, 3);

        long long BSK = (long long)Bb * S * Kn;
        int* knn = ws.i(BSK);
        knn_kernel<<<L1((long long)Bb * S), 256, 0, stream>>>(nxyz, xyz, knn, Bb, S, N, Kn);

        float* grouped = ws.f(BSK * C);
        gather_kernel<<<L1(BSK * C), 256, 0, stream>>>(feats, knn, grouped, Bb, N, S * Kn, C);

        float* h  = ws.f(BSK * C2);
        float* g2 = ws.f(BSK * C2);
        gemm(stream, ws, grouped, a.w,  a.b,  nullptr, h,  BSK, C2, C,  1);
        gemm(stream, ws, h,       b2.w, b2.b, nullptr, g2, BSK, C2, C2, 0);
        maxk_kernel<<<L1((long long)Bb * S * C2), 256, 0, stream>>>(g2, nfeats, Bb, S, Kn, C2);
        ws.off = mark;
    };

    // ================= forward =================
    int N = 1024;
    float* xyz = ws.f((size_t)Bb * N * 3);
    xyz_transpose_kernel<<<L1((long long)Bb * N * 3), 256, 0, stream>>>(X, xyz, Bb, N);

    float* feats = ws.f((size_t)Bb * N * 32);
    {
        size_t mark = ws.off;
        float* h0 = ws.f((size_t)Bb * N * 32);
        gemm(stream, ws, xyz, f1a.w, f1a.b, nullptr, h0,    (long long)Bb * N, 32, 3,  1);
        gemm(stream, ws, h0,  f1b.w, f1b.b, nullptr, feats, (long long)Bb * N, 32, 32, 0);
        ws.off = mark;
    }

    float *nx, *nf;
    tblock(t1, xyz, feats, N, 32);
    tdown(td1a, td1b, xyz, feats, N, 32, nx, nf);  xyz = nx; feats = nf; N /= 4;   // N=256, C=64
    tblock(t2, xyz, feats, N, 64);
    tdown(td2a, td2b, xyz, feats, N, 64, nx, nf);  xyz = nx; feats = nf; N /= 4;   // N=64, C=128
    tblock(t3, xyz, feats, N, 128);
    tdown(td3a, td3b, xyz, feats, N, 128, nx, nf); xyz = nx; feats = nf; N /= 4;   // N=16, C=256
    tblock(t4, xyz, feats, N, 256);
    tdown(td4a, td4b, xyz, feats, N, 256, nx, nf); xyz = nx; feats = nf; N /= 4;   // N=4, C=512
    tblock(t5, xyz, feats, N, 512);

    // ---- head ----
    float* g  = ws.f((size_t)Bb * 512);
    mean_kernel<<<L1((long long)Bb * 512), 256, 0, stream>>>(feats, g, Bb, N, 512);

    float* y1 = ws.f((size_t)Bb * 256);
    gemm(stream, ws, g, hl1.w, hl1.b, nullptr, y1, Bb, 256, 512, 0);
    bn_relu_kernel<<<L1((long long)Bb * 256), 256, 0, stream>>>(y1, hb1.g, hb1.b, hb1.rm, hb1.rv, Bb, 256);

    float* y2 = ws.f((size_t)Bb * 128);
    gemm(stream, ws, y1, hl2.w, hl2.b, nullptr, y2, Bb, 128, 256, 0);
    bn_relu_kernel<<<L1((long long)Bb * 128), 256, 0, stream>>>(y2, hb2.g, hb2.b, hb2.rm, hb2.rv, Bb, 128);

    gemm(stream, ws, y2, hl3.w, hl3.b, nullptr, (float*)d_out, Bb, 40, 128, 0);
}
